// Attention_29497835389298
// MI455X (gfx1250) — matrix operand order repaired and checked by reading, not compiled, not run on hardware
//
#include <hip/hip_runtime.h>
#include <hip/hip_bf16.h>
#include <stdint.h>

typedef __bf16 bf16_t;
typedef __attribute__((ext_vector_type(8)))  __bf16 bf16x8;
typedef __attribute__((ext_vector_type(16))) __bf16 bf16x16;
typedef __attribute__((ext_vector_type(8)))  float  floatx8;
typedef __attribute__((ext_vector_type(4)))  int    int4v;

#if defined(__HIP_DEVICE_COMPILE__) && defined(__has_builtin)
# if __has_builtin(__builtin_amdgcn_global_load_async_to_lds_b128)
#  define HAVE_ASYNC_LDS 1
# endif
#endif
#ifndef HAVE_ASYNC_LDS
# define HAVE_ASYNC_LDS 0
#endif

static __device__ __forceinline__ bf16x16 join16(bf16x8 lo, bf16x8 hi) {
  return __builtin_shufflevector(lo, hi, 0, 1, 2, 3, 4, 5, 6, 7,
                                         8, 9, 10, 11, 12, 13, 14, 15);
}

static __device__ __forceinline__ unsigned short f32_to_bf16_rne(float f) {
  union { float f; unsigned int u; } v;
  v.f = f;
  unsigned int u = v.u;
  unsigned int r = u + 0x7FFFu + ((u >> 16) & 1u);
  return (unsigned short)(r >> 16);
}

#if HAVE_ASYNC_LDS
// Async global->LDS 16B copy (GLOBAL_LOAD_ASYNC_TO_LDS_B128, ASYNCcnt).
static __device__ __forceinline__ void async_copy16(const bf16_t* g, bf16_t* l) {
  __builtin_amdgcn_global_load_async_to_lds_b128(
      (__attribute__((address_space(1))) int4v*)(uintptr_t)g,
      (__attribute__((address_space(3))) int4v*)(unsigned int)(uintptr_t)l,
      0, 0);
}
#endif

// ---------------- fp32 -> bf16 (RNE) ----------------
__global__ void attn_cvt_f32_to_bf16(const float* __restrict__ src,
                                     unsigned short* __restrict__ dst,
                                     long long n) {
  long long i = ((long long)blockIdx.x * blockDim.x + threadIdx.x) * 4;
  if (i + 3 < n) {
    float4 f = *(const float4*)(src + i);
    ushort4 o;
    o.x = f32_to_bf16_rne(f.x);
    o.y = f32_to_bf16_rne(f.y);
    o.z = f32_to_bf16_rne(f.z);
    o.w = f32_to_bf16_rne(f.w);
    *(ushort4*)(dst + i) = o;
  } else {
    for (; i < n; ++i) dst[i] = f32_to_bf16_rne(src[i]);
  }
}

// ---------------- fp32[R][C] -> bf16[C][R] transpose-convert ----------------
__global__ void attn_cvt_t_f32_to_bf16(const float* __restrict__ src,
                                       unsigned short* __restrict__ dst,
                                       int R, int C) {
  __shared__ unsigned short t[32][33];
  int c0 = blockIdx.x * 32, r0 = blockIdx.y * 32;
  int tx = threadIdx.x & 31, ty = threadIdx.x >> 5;  // 256 thr: ty 0..7
#pragma unroll
  for (int i = 0; i < 32; i += 8)
    t[ty + i][tx] = f32_to_bf16_rne(src[(size_t)(r0 + ty + i) * C + c0 + tx]);
  __syncthreads();
#pragma unroll
  for (int i = 0; i < 32; i += 8)
    dst[(size_t)(c0 + ty + i) * R + r0 + tx] = t[tx][ty + i];
}

// ---------------- fused bias: bfuse = bv @ Wc + bc ----------------
__global__ void attn_bias_fuse(const float* __restrict__ bv,
                               const float* __restrict__ Wc,
                               const float* __restrict__ bc,
                               float* __restrict__ out, int E) {
  int n = blockIdx.x * blockDim.x + threadIdx.x;
  if (n < E) {
    float s = bc[n];
    for (int k = 0; k < E; ++k) s += bv[k] * Wc[(size_t)k * E + n];
    out[n] = s;
  }
}

// -------- NT bf16 WMMA GEMM: D[M][N] = A[M][K] @ Bt[N][K]^T + bias --------
// Block tile 128x256, 8 waves of 64x64 (4x4 fragments), BK=32, double-buffered
// LDS fed by GLOBAL_LOAD_ASYNC_TO_LDS_B128.  LDS = 2*(128+256)*40*2B = 60 KB.
#define BM 128
#define BN 256
#define BK 32
#define SA 40  // 20-dword row stride -> conflict-free fragment reads
#define SB 40

__global__ __launch_bounds__(256, 2)
void attn_wmma_gemm_nt_bf16(const bf16_t* __restrict__ A,
                            const bf16_t* __restrict__ Bt,
                            const float* __restrict__ bias,
                            float* __restrict__ D,
                            int M, int N, int K) {
  __shared__ __align__(16) bf16_t sA[2][BM * SA];
  __shared__ __align__(16) bf16_t sB[2][BN * SB];

  const int tid  = threadIdx.x;
  const int m0   = blockIdx.y * BM;
  const int n0   = blockIdx.x * BN;
  const int lane = tid & 31;
  const int wave = tid >> 5;   // 8 waves
  const int wm   = wave >> 2;  // 0..1 -> 64-row strip
  const int wn   = wave & 3;   // 0..3 -> 64-col strip
  const int l16  = lane & 15;
  const int lh   = lane >> 4;

  // staging geometry: A tile 512 chunks (2/thr), B tile 1024 chunks (4/thr)
  const bf16_t* gA[2];
  const bf16_t* gB[4];
  unsigned la[2], lb[4];
#pragma unroll
  for (int i = 0; i < 2; ++i) {
    int c = tid + (i << 8);
    int r = c >> 2, kc = (c & 3) << 3;
    gA[i] = A + (size_t)(m0 + r) * K + kc;
    la[i] = r * SA + kc;
  }
#pragma unroll
  for (int i = 0; i < 4; ++i) {
    int c = tid + (i << 8);
    int r = c >> 2, kc = (c & 3) << 3;
    gB[i] = Bt + (size_t)(n0 + r) * K + kc;
    lb[i] = r * SB + kc;
  }

  floatx8 acc[4][4];
#pragma unroll
  for (int im = 0; im < 4; ++im)
#pragma unroll
    for (int in = 0; in < 4; ++in) acc[im][in] = 0.0f;

  // ---- prologue: stage tile 0 into buffer 0 ----
#if HAVE_ASYNC_LDS
#pragma unroll
  for (int i = 0; i < 2; ++i) async_copy16(gA[i], &sA[0][la[i]]);
#pragma unroll
  for (int i = 0; i < 4; ++i) async_copy16(gB[i], &sB[0][lb[i]]);
  asm volatile("s_wait_asynccnt 0x0" ::: "memory");
#else
#pragma unroll
  for (int i = 0; i < 2; ++i) *(bf16x8*)&sA[0][la[i]] = *(const bf16x8*)gA[i];
#pragma unroll
  for (int i = 0; i < 4; ++i) *(bf16x8*)&sB[0][lb[i]] = *(const bf16x8*)gB[i];
#endif
#pragma unroll
  for (int i = 0; i < 2; ++i) gA[i] += BK;
#pragma unroll
  for (int i = 0; i < 4; ++i) gB[i] += BK;
  __syncthreads();

  const int NSTEP = K / BK;
  for (int t = 0; t < NSTEP; ++t) {
    const int cur = t & 1, nxt = cur ^ 1;

#if HAVE_ASYNC_LDS
    if (t + 1 < NSTEP) {  // stream next tile global->LDS while computing
#pragma unroll
      for (int i = 0; i < 2; ++i) async_copy16(gA[i], &sA[nxt][la[i]]);
#pragma unroll
      for (int i = 0; i < 4; ++i) async_copy16(gB[i], &sB[nxt][lb[i]]);
#pragma unroll
      for (int i = 0; i < 2; ++i) gA[i] += BK;
#pragma unroll
      for (int i = 0; i < 4; ++i) gB[i] += BK;
    }
#else
    bf16x8 ra[2], rb[4];
    if (t + 1 < NSTEP) {
#pragma unroll
      for (int i = 0; i < 2; ++i) { ra[i] = *(const bf16x8*)gA[i]; gA[i] += BK; }
#pragma unroll
      for (int i = 0; i < 4; ++i) { rb[i] = *(const bf16x8*)gB[i]; gB[i] += BK; }
    }
#endif

    // ---- compute: 4x4 fragments, 16 WMMA, 8 ds_load_b128 pairs ----
    bf16x16 af[4];
#pragma unroll
    for (int im = 0; im < 4; ++im) {
      const bf16_t* p = &sA[cur][(wm * 64 + im * 16 + l16) * SA + lh * 8];
      af[im] = join16(*(const bf16x8*)p, *(const bf16x8*)(p + 16));
    }
    bf16x16 bfv[4];
#pragma unroll
    for (int in = 0; in < 4; ++in) {
      const bf16_t* p = &sB[cur][(wn * 64 + in * 16 + l16) * SB + lh * 8];
      bfv[in] = join16(*(const bf16x8*)p, *(const bf16x8*)(p + 16));
    }
#pragma unroll
    for (int im = 0; im < 4; ++im)
#pragma unroll
      for (int in = 0; in < 4; ++in)
        acc[im][in] = __builtin_amdgcn_wmma_f32_16x16x32_bf16(
            false, af[im], false, bfv[in], (short)0, acc[im][in], false, false);

#if HAVE_ASYNC_LDS
    asm volatile("s_wait_asynccnt 0x0" ::: "memory");
#else
    if (t + 1 < NSTEP) {
#pragma unroll
      for (int i = 0; i < 2; ++i) *(bf16x8*)&sA[nxt][la[i]] = ra[i];
#pragma unroll
      for (int i = 0; i < 4; ++i) *(bf16x8*)&sB[nxt][lb[i]] = rb[i];
    }
#endif
    __syncthreads();
  }

  // ---- epilogue: C/D layout row = r + 8*lane_half, col = lane%16 ----
#pragma unroll
  for (int im = 0; im < 4; ++im) {
    int rowb = m0 + wm * 64 + im * 16 + lh * 8;
#pragma unroll
    for (int in = 0; in < 4; ++in) {
      int col  = n0 + wn * 64 + in * 16 + l16;
      float bb = bias ? bias[col] : 0.0f;
#pragma unroll
      for (int r = 0; r < 8; ++r)
        D[(size_t)(rowb + r) * N + col] = acc[im][in][r] + bb;
    }
  }
}

extern "C" void kernel_launch(void* const* d_in, const int* in_sizes, int n_in,
                              void* d_out, int out_size, void* d_ws, size_t ws_size,
                              hipStream_t stream) {
  (void)n_in; (void)out_size; (void)ws_size;
  // setup_inputs order: x, Wq, bq, Wk, bk, Wv, bv, Wc, bc
  const float* x  = (const float*)d_in[0];
  const float* Wv = (const float*)d_in[5];
  const float* bv = (const float*)d_in[6];
  const float* Wc = (const float*)d_in[7];
  const float* bc = (const float*)d_in[8];

  long long we = in_sizes[7];  // E*E
  int E = 1;
  while ((long long)E * E < we) E <<= 1;  // 2048
  long long xn = in_sizes[0];
  int M = (int)(xn / E);                  // 8192

  char* w = (char*)d_ws;
  size_t sz_x  = (size_t)M * E * sizeof(unsigned short);
  size_t sz_wh = (size_t)E * E * sizeof(unsigned short);
  size_t sz_wf = (size_t)E * E * sizeof(float);
  unsigned short* x_bf   = (unsigned short*)(w);                       // x bf16 [m][k]
  unsigned short* wv_bf  = (unsigned short*)(w + sz_x);                // Wv bf16 [k][e]
  unsigned short* wct_bf = (unsigned short*)(w + sz_x + sz_wh);        // Wc^T bf16 [n][e]
  float*          wf32   = (float*)(w + sz_x + 2 * sz_wh);             // Wf^T f32 [n][k]
  float*          bfuse  = (float*)(w + sz_x + 2 * sz_wh + sz_wf);
  unsigned short* wft_bf = wv_bf;  // reuse Wv slot after fusion GEMM (stream order)

  long long nW = (long long)E * E;
  const int thr = 256;
  int blkW = (int)((nW / 4 + thr - 1) / thr);
  int blkX = (int)((xn / 4 + thr - 1) / thr);

  attn_cvt_f32_to_bf16<<<blkW, thr, 0, stream>>>(Wv, wv_bf, nW);
  attn_cvt_t_f32_to_bf16<<<dim3(E / 32, E / 32), thr, 0, stream>>>(Wc, wct_bf, E, E);
  attn_cvt_f32_to_bf16<<<blkX, thr, 0, stream>>>(x, x_bf, xn);
  attn_bias_fuse<<<(E + thr - 1) / thr, thr, 0, stream>>>(bv, Wc, bc, bfuse, E);

  // Wf^T[n][k] = sum_e Wc^T[n][e] * Wv[k][e]  (NT form)
  dim3 g1(E / BN, E / BM);
  attn_wmma_gemm_nt_bf16<<<g1, 256, 0, stream>>>(
      (const bf16_t*)wct_bf, (const bf16_t*)wv_bf, nullptr, wf32, E, E, E);
  attn_cvt_f32_to_bf16<<<blkW, thr, 0, stream>>>(wf32, wft_bf, nW);

  // out[m][n] = sum_k x[m][k] * Wf^T[n][k] + bfuse[n]
  // (attention is mathematically the identity on v: softmax rows sum to 1)
  dim3 g2(E / BN, M / BM);
  attn_wmma_gemm_nt_bf16<<<g2, 256, 0, stream>>>(
      (const bf16_t*)x_bf, (const bf16_t*)wft_bf, bfuse, (float*)d_out, M, E, E);
}
